// GATv2ConvNN_86938728005830
// MI455X (gfx1250) — matrix operand order repaired and checked by reading, not compiled, not run on hardware
//
#include <hip/hip_runtime.h>
#include <hip/hip_bf16.h>

typedef __attribute__((ext_vector_type(16))) _Float16 v16h;
typedef __attribute__((ext_vector_type(8)))  _Float16 v8h;
typedef __attribute__((ext_vector_type(8)))  float    v8f;
typedef __attribute__((ext_vector_type(4)))  float    v4f;

#define N_NODES 50000
#define N_EDGES 1200000
#define E_TOTAL (N_EDGES + N_NODES)
#define IN_CH   128
#define HID     64
#define OUT_CH  64
#define M_TILES ((N_NODES + 15) / 16)        // 3125 (exact: 50000 = 3125*16)
#define TILE_BLOCKS ((M_TILES + 1) / 2)      // 1563

// ---- monotone float <-> uint key for atomicMax-based segment max ----
__device__ __forceinline__ unsigned fkey(float f) {
    int i = __float_as_int(f);
    return (unsigned)(i ^ ((i >> 31) | 0x80000000));
}
__device__ __forceinline__ float funkey(unsigned u) {
    int i = (int)(u ^ (((int)u >= 0) ? 0xFFFFFFFFu : 0x80000000u));
    return __int_as_float(i);
}

// ------------------- 1) init scratch -------------------
__global__ __launch_bounds__(256) void k_init(float* __restrict__ agg,
                                              unsigned* __restrict__ mkeys,
                                              float* __restrict__ denom) {
    int t = blockIdx.x * 256 + threadIdx.x;
    if (t < N_NODES * HID) agg[t] = 0.0f;
    if (t < N_NODES) {
        mkeys[t] = fkey(-__builtin_inff());
        denom[t] = 0.0f;
    }
}

// ------------- 2) fused WMMA projections: xl = x@W_l^T AND xr = x@W_r^T ----
// x read once (f32 -> f16 inline). grid = TILE_BLOCKS; block = 256 (8 waves).
// Each wave: one 16x16 M/N tile, TWO accumulators (l and r), K = 4 x 32.
__global__ __launch_bounds__(256) void k_proj(const float* __restrict__ x,
                                              const float* __restrict__ Wl,
                                              const float* __restrict__ Wr,
                                              float* __restrict__ xl,
                                              float* __restrict__ xr) {
    __shared__ alignas(32) _Float16 WtL[IN_CH * HID];  // W_l f16: [n][k], 16 KB
    __shared__ alignas(32) _Float16 WtR[IN_CH * HID];  // W_r f16: [n][k], 16 KB

    int tid = threadIdx.x;
    for (int i = tid; i < HID * IN_CH; i += 256) {     // W row-major [n][k]
        int n = i >> 7, k = i & (IN_CH - 1);
        WtL[n * IN_CH + k] = (_Float16)Wl[i];
        WtR[n * IN_CH + k] = (_Float16)Wr[i];
    }
    __syncthreads();

    int wave  = tid >> 5, lane = tid & 31;
    int mtile = blockIdx.x * 2 + (wave >> 2);
    int ntile = wave & 3;
    int khalf = lane >> 4;
    int grow  = mtile * 16 + (lane & 15);
    int arow  = grow < N_NODES ? grow : N_NODES - 1;   // clamp; keep full EXEC

    v8f cl = {}, cr = {};
#pragma unroll
    for (int kk = 0; kk < 4; ++kk) {                   // K = 4 x 32
        int bk = kk * 32 + khalf * 8;
        const float* ap = x + (size_t)arow * IN_CH + bk;
        v4f p0 = *(const v4f*)(ap);
        v4f p1 = *(const v4f*)(ap + 4);
        v4f p2 = *(const v4f*)(ap + 16);
        v4f p3 = *(const v4f*)(ap + 20);
        union { v16h v; _Float16 e[16]; } a;
#pragma unroll
        for (int j = 0; j < 4; ++j) {
            a.e[j]      = (_Float16)p0[j];
            a.e[4 + j]  = (_Float16)p1[j];
            a.e[8 + j]  = (_Float16)p2[j];
            a.e[12 + j] = (_Float16)p3[j];
        }
        int brow = (ntile * 16 + (lane & 15)) * IN_CH + bk;
        v16h bl = __builtin_shufflevector(*(const v8h*)(WtL + brow), *(const v8h*)(WtL + brow + 16), 0,1,2,3,4,5,6,7,8,9,10,11,12,13,14,15);
        v16h br = __builtin_shufflevector(*(const v8h*)(WtR + brow), *(const v8h*)(WtR + brow + 16), 0,1,2,3,4,5,6,7,8,9,10,11,12,13,14,15);
        cl = __builtin_amdgcn_wmma_f32_16x16x32_f16(false, a.v, false, bl,
                                                    (short)0, cl, false, false);
        cr = __builtin_amdgcn_wmma_f32_16x16x32_f16(false, a.v, false, br,
                                                    (short)0, cr, false, false);
    }

    if (mtile < M_TILES) {                             // wave-uniform: tiles are full
        int col = ntile * 16 + (lane & 15);
        size_t base = (size_t)(mtile * 16 + khalf * 8) * HID + col;
#pragma unroll
        for (int e = 0; e < 8; ++e) {
            xl[base + (size_t)e * HID] = cl[e];
            xr[base + (size_t)e * HID] = cr[e];
        }
    }
}

// ------------------- 3) wave-per-edge logits + segment max ----------------
__global__ __launch_bounds__(256) void k_logits(const long long* __restrict__ ei,
                                                const float* __restrict__ xl,
                                                const float* __restrict__ xr,
                                                const float* __restrict__ att,
                                                float* __restrict__ logits,
                                                unsigned* __restrict__ mkeys) {
    int wid  = (blockIdx.x * 256 + threadIdx.x) >> 5;
    int lane = threadIdx.x & 31;
    if (wid >= E_TOTAL) return;
    long long s, d;
    if (wid < N_EDGES) { s = ei[wid]; d = ei[N_EDGES + wid]; }
    else               { s = d = wid - N_EDGES; }

    float acc = 0.0f;
#pragma unroll
    for (int h = 0; h < 2; ++h) {
        int c = lane + h * 32;
        float v = xl[(size_t)s * HID + c] + xr[(size_t)d * HID + c];
        v = v > 0.0f ? v : 0.2f * v;                   // LeakyReLU(0.2)
        acc += att[c] * v;
    }
#pragma unroll
    for (int off = 16; off; off >>= 1) acc += __shfl_xor(acc, off);
    if (lane == 0) {
        logits[wid] = acc;
        atomicMax(&mkeys[d], fkey(acc));
    }
}

// ------------------- 4) exp(logit - max) + segment sum --------------------
__global__ __launch_bounds__(256) void k_expsum(const long long* __restrict__ ei,
                                                const unsigned* __restrict__ mkeys,
                                                float* __restrict__ logits,
                                                float* __restrict__ denom) {
    int e = blockIdx.x * 256 + threadIdx.x;
    if (e >= E_TOTAL) return;
    long long d = (e < N_EDGES) ? ei[N_EDGES + e] : (long long)(e - N_EDGES);
    float m  = funkey(mkeys[d]);
    float ex = __expf(logits[e] - m);
    logits[e] = ex;                                    // reuse buffer as expv
    atomicAdd(&denom[d], ex);
}

// ------------------- 5) wave-per-edge weighted scatter --------------------
__global__ __launch_bounds__(256) void k_agg(const long long* __restrict__ ei,
                                             const float* __restrict__ xl,
                                             const float* __restrict__ expv,
                                             const float* __restrict__ denom,
                                             float* __restrict__ agg) {
    int wid  = (blockIdx.x * 256 + threadIdx.x) >> 5;
    int lane = threadIdx.x & 31;
    if (wid >= E_TOTAL) return;
    long long s, d;
    if (wid < N_EDGES) { s = ei[wid]; d = ei[N_EDGES + wid]; }
    else               { s = d = wid - N_EDGES; }
    float alpha = expv[wid] / (denom[d] + 1e-16f);
#pragma unroll
    for (int h = 0; h < 2; ++h) {
        int c = lane + h * 32;
        atomicAdd(&agg[(size_t)d * HID + c], alpha * xl[(size_t)s * HID + c]);
    }
}

// ------------- 6) out = ELU(agg + bias) @ W_lin^T + b_lin (WMMA) ----------
__device__ __forceinline__ float eluf(float v) {
    return v > 0.0f ? v : (__expf(v) - 1.0f);
}

__global__ __launch_bounds__(256) void k_final(const float* __restrict__ agg,
                                               const float* __restrict__ bias_conv,
                                               const float* __restrict__ Wlin,
                                               const float* __restrict__ blin,
                                               float* __restrict__ out) {
    __shared__ alignas(32) _Float16 Wt[HID * OUT_CH];  // [n][k], 8 KB
    int tid = threadIdx.x;
    for (int i = tid; i < OUT_CH * HID; i += 256) {    // W_lin row-major [n][k]
        int n = i >> 6, k = i & (HID - 1);
        Wt[n * HID + k] = (_Float16)Wlin[i];
    }
    __syncthreads();

    int wave  = tid >> 5, lane = tid & 31;
    int mtile = blockIdx.x * 2 + (wave >> 2);
    int ntile = wave & 3;
    int khalf = lane >> 4;
    int grow  = mtile * 16 + (lane & 15);
    int arow  = grow < N_NODES ? grow : N_NODES - 1;

    v8f c = {};
#pragma unroll
    for (int kk = 0; kk < 2; ++kk) {                   // K = 2 x 32
        int bk = kk * 32 + khalf * 8;
        const float* ap = agg + (size_t)arow * HID + bk;
        v4f p0 = *(const v4f*)(ap);
        v4f p1 = *(const v4f*)(ap + 4);
        v4f p2 = *(const v4f*)(ap + 16);
        v4f p3 = *(const v4f*)(ap + 20);
        union { v16h v; _Float16 e[16]; } a;
#pragma unroll
        for (int j = 0; j < 4; ++j) {
            a.e[j]      = (_Float16)eluf(p0[j] + bias_conv[bk + j]);
            a.e[4 + j]  = (_Float16)eluf(p1[j] + bias_conv[bk + 4 + j]);
            a.e[8 + j]  = (_Float16)eluf(p2[j] + bias_conv[bk + 16 + j]);
            a.e[12 + j] = (_Float16)eluf(p3[j] + bias_conv[bk + 20 + j]);
        }
        int brow = (ntile * 16 + (lane & 15)) * HID + bk;
        v16h b = __builtin_shufflevector(*(const v8h*)(Wt + brow), *(const v8h*)(Wt + brow + 16), 0,1,2,3,4,5,6,7,8,9,10,11,12,13,14,15);
        c = __builtin_amdgcn_wmma_f32_16x16x32_f16(false, a.v, false, b,
                                                   (short)0, c, false, false);
    }

    if (mtile < M_TILES) {                             // wave-uniform: tiles are full
        int col = ntile * 16 + (lane & 15);
        float bv = blin[col];
        size_t base = (size_t)(mtile * 16 + khalf * 8) * OUT_CH + col;
#pragma unroll
        for (int e = 0; e < 8; ++e)
            out[base + (size_t)e * OUT_CH] = c[e] + bv;
    }
}

extern "C" void kernel_launch(void* const* d_in, const int* in_sizes, int n_in,
                              void* d_out, int out_size, void* d_ws, size_t ws_size,
                              hipStream_t stream) {
    const float*     x         = (const float*)d_in[0];
    const long long* ei        = (const long long*)d_in[1];   // int64 [2, E]
    // d_in[2] = edge_weight: unused by the reference
    const float*     W_l       = (const float*)d_in[3];
    const float*     W_r       = (const float*)d_in[4];
    const float*     att       = (const float*)d_in[5];
    const float*     bias_conv = (const float*)d_in[6];
    const float*     W_lin     = (const float*)d_in[7];
    const float*     b_lin     = (const float*)d_in[8];
    float*           out       = (float*)d_out;

    // ---- workspace layout (~44 MB) ----
    char* ws = (char*)d_ws;
    float*    xl     = (float*)(ws);                             // 12.8 MB
    float*    xr     = (float*)(ws + 12800000);                  // 12.8 MB
    float*    agg    = (float*)(ws + 25600000);                  // 12.8 MB
    float*    logits = (float*)(ws + 38400000);                  // 5.0 MB
    unsigned* mkeys  = (unsigned*)(ws + 43400000);               // 0.2 MB
    float*    denom  = (float*)(ws + 43600000);                  // 0.2 MB

    (void)in_sizes; (void)n_in; (void)out_size; (void)ws_size;

    k_init<<<(N_NODES * HID + 255) / 256, 256, 0, stream>>>(agg, mkeys, denom);
    k_proj<<<TILE_BLOCKS, 256, 0, stream>>>(x, W_l, W_r, xl, xr);
    {
        int blocks = (E_TOTAL * 32 + 255) / 256;       // wave per edge
        k_logits<<<blocks, 256, 0, stream>>>(ei, xl, xr, att, logits, mkeys);
        k_expsum<<<(E_TOTAL + 255) / 256, 256, 0, stream>>>(ei, mkeys, logits, denom);
        k_agg<<<blocks, 256, 0, stream>>>(ei, xl, logits, denom, agg);
    }
    k_final<<<TILE_BLOCKS, 256, 0, stream>>>(agg, bias_conv, W_lin, b_lin, out);
}
